// graph_att_submodel_76158360092907
// MI455X (gfx1250) — matrix operand order repaired and checked by reading, not compiled, not run on hardware
//
#include <hip/hip_runtime.h>
#include <hip/hip_bf16.h>
#include <math.h>

#define B_ 16
#define S_ 128
#define R_ 32
#define C_ 64
#define D_ 256
#define H_ 512
#define L_ 2
#define NEGV (-10000.0f)

typedef __attribute__((ext_vector_type(16))) __bf16 v16bf;
typedef __attribute__((ext_vector_type(8)))  float  v8f;

__device__ __forceinline__ unsigned short f2bf(float f) {
  unsigned int u = __float_as_uint(f);
  if ((u & 0x7F800000u) == 0x7F800000u && (u & 0x007FFFFFu))
    return (unsigned short)((u >> 16) | 0x40u);              // quiet the NaN
  unsigned int r = u + 0x7FFFu + ((u >> 16) & 1u);           // round-nearest-even
  return (unsigned short)(r >> 16);
}

// ---------------------------------------------------------------------------
// f32 -> bf16 conversion (weights, activations)
// ---------------------------------------------------------------------------
__global__ void k_f2bf(const float* __restrict__ in, unsigned short* __restrict__ out, int n) {
  int i = blockIdx.x * 256 + threadIdx.x;
  if (i < n) out[i] = f2bf(in[i]);
}

// ---------------------------------------------------------------------------
// bf16 WMMA GEMM: Out[M,N] = act(A[M,K] @ W[K,N] + bias) (+ residual)
// BM=64, BN=64, BK=32. 256 threads = 8 waves; wave tile = 16x32 (2 wmma/step).
// A tile:  async global->LDS (ASYNCcnt), row-major padded stride 40 ushorts.
// B tile:  global->reg, v_perm transpose-pack to Bt[n][k/2] u32 words.
// Fragments: 2 x ds_load_b128 each. Double-buffered, one barrier per k-step.
// ---------------------------------------------------------------------------
template <int K>
__global__ __launch_bounds__(256) void k_gemm_bf16(
    const unsigned short* __restrict__ A, int lda,
    const unsigned short* __restrict__ W, int ldb,
    const float* __restrict__ bias,
    const float* __restrict__ residual,
    float* __restrict__ outF,
    unsigned short* __restrict__ outH,
    int N, int act)
{
  __shared__ __align__(16) unsigned short As[2][64 * 40];
  __shared__ __align__(16) unsigned int   Bt[2][64 * 20];

  const int tid  = threadIdx.x;
  const int bM   = blockIdx.x * 64;
  const int bN   = blockIdx.y * 64;
  const int wave = tid >> 5;
  const int lane = tid & 31;
  const int lm   = lane & 15;
  const int lh   = lane >> 4;
  const int m0   = (wave & 3) * 16;
  const int n0   = (wave >> 2) * 32;

  v8f acc0 = {0.f, 0.f, 0.f, 0.f, 0.f, 0.f, 0.f, 0.f};
  v8f acc1 = {0.f, 0.f, 0.f, 0.f, 0.f, 0.f, 0.f, 0.f};

  // staging assignments
  const int arow = tid >> 2;          // 0..63
  const int acg  = (tid & 3) * 8;     // 0,8,16,24 (16B chunks)
  const int bkp  = tid >> 3;          // k-pair 0..15 (tid<128)
  const int bng  = (tid & 7) * 8;     // n group

  constexpr int NK = K >> 5;

  const unsigned short* aSrc = &A[(size_t)(bM + arow) * lda + acg];
  const unsigned aDst[2] = {
    (unsigned)(size_t)&As[0][arow * 40 + acg],
    (unsigned)(size_t)&As[1][arow * 40 + acg]
  };

  // prologue: tile 0
  asm volatile("global_load_async_to_lds_b128 %0, %1, off"
               :: "v"(aDst[0]), "v"(aSrc) : "memory");
  uint4 b0Reg = {}, b1Reg = {};
  if (tid < 128) {
    b0Reg = *(const uint4*)&W[(size_t)(2 * bkp) * ldb + bN + bng];
    b1Reg = *(const uint4*)&W[(size_t)(2 * bkp + 1) * ldb + bN + bng];
    const unsigned* q0 = (const unsigned*)&b0Reg;
    const unsigned* q1 = (const unsigned*)&b1Reg;
    #pragma unroll
    for (int t = 0; t < 4; ++t) {
      Bt[0][(bng + 2 * t)     * 20 + bkp] = __builtin_amdgcn_perm(q1[t], q0[t], 0x05040100u);
      Bt[0][(bng + 2 * t + 1) * 20 + bkp] = __builtin_amdgcn_perm(q1[t], q0[t], 0x07060302u);
    }
  }
  asm volatile("s_wait_asynccnt 0x0" ::: "memory");
  __syncthreads();

  int buf = 0;
  #pragma unroll 2
  for (int kt = 0; kt < NK; ++kt) {
    const bool more = (kt + 1) < NK;
    if (more) {   // prefetch next tile: async A -> other LDS buffer, B -> regs
      const int k0 = (kt + 1) << 5;
      asm volatile("global_load_async_to_lds_b128 %0, %1, off"
                   :: "v"(aDst[buf ^ 1]), "v"(aSrc + k0) : "memory");
      if (tid < 128) {
        b0Reg = *(const uint4*)&W[(size_t)(k0 + 2 * bkp) * ldb + bN + bng];
        b1Reg = *(const uint4*)&W[(size_t)(k0 + 2 * bkp + 1) * ldb + bN + bng];
      }
    }
    // A fragment: 2 aligned b128 LDS loads
    union { v16bf v; uint4 q[2]; } af;
    const unsigned short* ar = &As[buf][(m0 + lm) * 40];
    af.q[0] = *(const uint4*)&ar[16 * lh];
    af.q[1] = *(const uint4*)&ar[16 * lh + 8];
    // B fragments: 2 aligned b128 LDS loads each
    union { v16bf v; uint4 q[2]; } bf0, bf1;
    const unsigned int* b0p = &Bt[buf][(n0 + lm) * 20 + 8 * lh];
    bf0.q[0] = *(const uint4*)&b0p[0];
    bf0.q[1] = *(const uint4*)&b0p[4];
    const unsigned int* b1p = &Bt[buf][(n0 + 16 + lm) * 20 + 8 * lh];
    bf1.q[0] = *(const uint4*)&b1p[0];
    bf1.q[1] = *(const uint4*)&b1p[4];

    acc0 = __builtin_amdgcn_wmma_f32_16x16x32_bf16(false, af.v, false, bf0.v,
                                                   (short)0, acc0, false, false);
    acc1 = __builtin_amdgcn_wmma_f32_16x16x32_bf16(false, af.v, false, bf1.v,
                                                   (short)0, acc1, false, false);

    if (more) {   // commit B regs via v_perm pack; join async A; barrier
      if (tid < 128) {
        const unsigned* q0 = (const unsigned*)&b0Reg;
        const unsigned* q1 = (const unsigned*)&b1Reg;
        #pragma unroll
        for (int t = 0; t < 4; ++t) {
          Bt[buf ^ 1][(bng + 2 * t)     * 20 + bkp] = __builtin_amdgcn_perm(q1[t], q0[t], 0x05040100u);
          Bt[buf ^ 1][(bng + 2 * t + 1) * 20 + bkp] = __builtin_amdgcn_perm(q1[t], q0[t], 0x07060302u);
        }
      }
      asm volatile("s_wait_asynccnt 0x0" ::: "memory");
      __syncthreads();
      buf ^= 1;
    }
  }

  // C/D layout: lane half lh -> rows m0+8*lh+r, column n0+lm (acc1: +16)
  const int gm = bM + m0 + 8 * lh;
  #pragma unroll
  for (int half = 0; half < 2; ++half) {
    const v8f& acc = half ? acc1 : acc0;
    const int gn = bN + n0 + half * 16 + lm;
    const float bia = bias ? bias[gn] : 0.f;
    #pragma unroll
    for (int r = 0; r < 8; ++r) {
      float v = acc[r] + bia;
      if (act) v = tanhf(v);
      size_t idx = (size_t)(gm + r) * N + gn;
      if (residual) v += residual[idx];
      if (outF) outF[idx] = v;
      if (outH) outH[idx] = f2bf(v);
    }
  }
}

// ---------------------------------------------------------------------------
// Masked-softmax pooling: out[b,n,:] = softmax(where(mask,NEG,att)) @ lstm[b]
// ---------------------------------------------------------------------------
__global__ __launch_bounds__(128) void k_pooled(
    const float* __restrict__ att, const float* __restrict__ lstm,
    const int* __restrict__ mask, float* __restrict__ outF,
    unsigned short* __restrict__ outH, int Ncols)
{
  __shared__ float a[S_];
  __shared__ float red[S_];
  const int bn = blockIdx.x;
  const int b  = bn / Ncols;
  const int j  = threadIdx.x;
  const float s = mask[(size_t)bn * S_ + j] ? NEGV : att[b * S_ + j];
  red[j] = s; __syncthreads();
  for (int off = 64; off > 0; off >>= 1) { if (j < off) red[j] = fmaxf(red[j], red[j + off]); __syncthreads(); }
  const float mx = red[0]; __syncthreads();
  const float e = __expf(s - mx);
  red[j] = e; __syncthreads();
  for (int off = 64; off > 0; off >>= 1) { if (j < off) red[j] += red[j + off]; __syncthreads(); }
  const float inv = 1.f / red[0];
  a[j] = e * inv; __syncthreads();
  #pragma unroll
  for (int dd = 0; dd < 2; ++dd) {
    const int d = j + dd * 128;
    float accv = 0.f;
    for (int jj = 0; jj < S_; ++jj) accv += a[jj] * lstm[(size_t)(b * S_ + jj) * D_ + d];
    outF[(size_t)bn * D_ + d] = accv;
    if (outH) outH[(size_t)bn * D_ + d] = f2bf(accv);
  }
}

// edge_vecs gather into second half of child_edge (bf16)
__global__ void k_edge_fill(const int* __restrict__ labels,
                            const unsigned short* __restrict__ embbf,
                            unsigned short* __restrict__ child_edge) {
  int row = blockIdx.x;                 // b*C+c
  int lab = labels[row];
  int d = threadIdx.x;
  child_edge[(size_t)row * 2 * D_ + D_ + d] = embbf[(size_t)lab * D_ + d];
}

// refresh first half of child_edge from current child_vecs
__global__ void k_child_refresh(const float* __restrict__ child,
                                unsigned short* __restrict__ child_edge) {
  int row = blockIdx.x;
  int d = threadIdx.x;
  child_edge[(size_t)row * 2 * D_ + d] = f2bf(child[(size_t)row * D_ + d]);
}

// GC_in = concat(root[gatt_root_idx], child_edge)  (bf16 [B*C, 3D])
__global__ void k_gcin(const unsigned short* __restrict__ rootbf,
                       const unsigned short* __restrict__ child_edge,
                       const int* __restrict__ gidx,
                       unsigned short* __restrict__ gcin) {
  int row = blockIdx.x;                 // b*C+c
  int b = row / C_;
  int r = gidx[row];
  int d = threadIdx.x;
  gcin[(size_t)row * 3 * D_ + d]            = rootbf[(size_t)(b * R_ + r) * D_ + d];
  gcin[(size_t)row * 3 * D_ + D_ + d]       = child_edge[(size_t)row * 2 * D_ + d];
  gcin[(size_t)row * 3 * D_ + 2 * D_ + d]   = child_edge[(size_t)row * 2 * D_ + D_ + d];
}

// HP[b,r,c,:] = bf16(tanh(Rpart[b,r,:] + CEpart[b,c,:]))
__global__ void k_hp(const float* __restrict__ Rpart, const float* __restrict__ CEpart,
                     unsigned short* __restrict__ HP) {
  int row = blockIdx.x;                 // br*C + c
  int c  = row & (C_ - 1);
  int br = row >> 6;
  int b  = br >> 5;
  int ceRow = b * C_ + c;
  #pragma unroll
  for (int it = 0; it < 2; ++it) {
    int hh = threadIdx.x + it * 256;
    float v = Rpart[(size_t)br * H_ + hh] + CEpart[(size_t)ceRow * H_ + hh];
    HP[(size_t)row * H_ + hh] = f2bf(tanhf(v));
  }
}

// graph attention: scores = mask? NEG : dot(E+RA, H); softmax over C; g = aw @ H
__global__ __launch_bounds__(256) void k_gatt(
    const float* __restrict__ Hf, const float* __restrict__ Ef,
    const float* __restrict__ RA, const int* __restrict__ gmask,
    float* __restrict__ grootF, unsigned short* __restrict__ grootH)
{
  __shared__ float part[256];
  __shared__ float sc[C_];
  const int br = blockIdx.x;            // b*R + r
  const int t  = threadIdx.x;
  const int c  = t >> 2, q = t & 3;
  const float* Hrow = Hf + (size_t)(br * C_ + c) * D_;
  const float* Erow = Ef + (size_t)(br * C_ + c) * D_;
  const float* RAr  = RA + (size_t)br * D_;
  float accv = 0.f;
  for (int d = q * 64; d < q * 64 + 64; ++d) accv += (Erow[d] + RAr[d]) * Hrow[d];
  part[t] = accv; __syncthreads();
  if (q == 0) {
    float sv = part[t] + part[t + 1] + part[t + 2] + part[t + 3];
    sc[c] = gmask[(size_t)br * C_ + c] ? NEGV : sv;
  }
  __syncthreads();
  if (t == 0) {
    float mx = sc[0];
    for (int i = 1; i < C_; ++i) mx = fmaxf(mx, sc[i]);
    float sum = 0.f;
    for (int i = 0; i < C_; ++i) { float e = __expf(sc[i] - mx); sc[i] = e; sum += e; }
    float inv = 1.f / sum;
    for (int i = 0; i < C_; ++i) sc[i] *= inv;
  }
  __syncthreads();
  float g = 0.f;
  for (int cc = 0; cc < C_; ++cc) g += sc[cc] * Hf[(size_t)(br * C_ + cc) * D_ + t];
  grootF[(size_t)br * D_ + t] = g;
  grootH[(size_t)br * D_ + t] = f2bf(g);
}

// final: s = mask? NEG : dot(fa, root); softmax over R; out = w @ root
__global__ __launch_bounds__(256) void k_final(
    const float* __restrict__ FA, const float* __restrict__ root,
    const int* __restrict__ smask, float* __restrict__ outp)
{
  __shared__ float part[256];
  __shared__ float sw[R_];
  const int b = blockIdx.x;
  const int t = threadIdx.x;
  const int r = t >> 3, q = t & 7;
  const float* far = FA + (size_t)(b * R_ + r) * D_;
  const float* rv  = root + (size_t)(b * R_ + r) * D_;
  float accv = 0.f;
  for (int d = q * 32; d < q * 32 + 32; ++d) accv += far[d] * rv[d];
  part[t] = accv; __syncthreads();
  if (q == 0) {
    float sv = 0.f;
    #pragma unroll
    for (int i = 0; i < 8; ++i) sv += part[t + i];
    sw[r] = smask[b * R_ + r] ? NEGV : sv;
  }
  __syncthreads();
  if (t == 0) {
    float mx = sw[0];
    for (int i = 1; i < R_; ++i) mx = fmaxf(mx, sw[i]);
    float sum = 0.f;
    for (int i = 0; i < R_; ++i) { float e = __expf(sw[i] - mx); sw[i] = e; sum += e; }
    float inv = 1.f / sum;
    for (int i = 0; i < R_; ++i) sw[i] *= inv;
  }
  __syncthreads();
  float o = 0.f;
  for (int rr = 0; rr < R_; ++rr) o += sw[rr] * root[(size_t)(b * R_ + rr) * D_ + t];
  outp[b * D_ + t] = o;
}

// ---------------------------------------------------------------------------
extern "C" void kernel_launch(void* const* d_in, const int* in_sizes, int n_in,
                              void* d_out, int out_size, void* d_ws, size_t ws_size,
                              hipStream_t stream)
{
  const float* lstm = (const float*)d_in[0];
  const float* att  = (const float*)d_in[1];
  const float* emb  = (const float*)d_in[2];
  const float* W1r  = (const float*)d_in[3];
  const float* b1r  = (const float*)d_in[4];
  const float* W2r  = (const float*)d_in[5];
  const float* b2r  = (const float*)d_in[6];
  const float* Wa   = (const float*)d_in[7];
  const float* ba   = (const float*)d_in[8];
  const float* W1c  = (const float*)d_in[9];
  const float* b1c  = (const float*)d_in[10];
  const float* W2c  = (const float*)d_in[11];
  const float* b2c  = (const float*)d_in[12];
  const float* Wf   = (const float*)d_in[13];
  const float* bfv  = (const float*)d_in[14];
  const float* Wfa  = (const float*)d_in[15];
  const float* bfa  = (const float*)d_in[16];
  const int* labels = (const int*)d_in[17];
  const int* rmask  = (const int*)d_in[18];
  const int* cmask  = (const int*)d_in[19];
  const int* gmask  = (const int*)d_in[20];
  const int* gidx   = (const int*)d_in[21];
  const int* smask  = (const int*)d_in[22];
  float* outp = (float*)d_out;
  (void)in_sizes; (void)n_in; (void)out_size; (void)ws_size;

  char* ws = (char*)d_ws;
  size_t cur = 0;
  auto alloc = [&](size_t bytes) -> void* {
    void* p = ws + cur;
    cur = (cur + bytes + 255) & ~((size_t)255);
    return p;
  };

  // bf16 weight copies
  unsigned short* emb_bf = (unsigned short*)alloc((size_t)50 * D_ * 2);
  unsigned short* W1r_bf = (unsigned short*)alloc((size_t)L_ * 3 * D_ * H_ * 2);
  unsigned short* W2r_bf = (unsigned short*)alloc((size_t)L_ * H_ * D_ * 2);
  unsigned short* Wa_bf  = (unsigned short*)alloc((size_t)L_ * 2 * D_ * D_ * 2);
  unsigned short* W1c_bf = (unsigned short*)alloc((size_t)L_ * 3 * D_ * H_ * 2);
  unsigned short* W2c_bf = (unsigned short*)alloc((size_t)L_ * H_ * D_ * 2);
  unsigned short* Wf_bf  = (unsigned short*)alloc((size_t)L_ * D_ * D_ * 2);
  unsigned short* Wfa_bf = (unsigned short*)alloc((size_t)D_ * D_ * 2);
  // activations
  float*          root_f   = (float*)alloc((size_t)B_ * R_ * D_ * 4);
  unsigned short* root_bf  = (unsigned short*)alloc((size_t)B_ * R_ * D_ * 2);
  float*          child_f  = (float*)alloc((size_t)B_ * C_ * D_ * 4);
  unsigned short* ce_bf    = (unsigned short*)alloc((size_t)B_ * C_ * 2 * D_ * 2);
  unsigned short* gcin_bf  = (unsigned short*)alloc((size_t)B_ * C_ * 3 * D_ * 2);
  unsigned short* T1_bf    = (unsigned short*)alloc((size_t)B_ * C_ * H_ * 2);
  unsigned short* GC_bf    = (unsigned short*)alloc((size_t)B_ * C_ * D_ * 2);
  float*          Rpart    = (float*)alloc((size_t)B_ * R_ * H_ * 4);
  float*          CEpart   = (float*)alloc((size_t)B_ * C_ * H_ * 4);
  float*          RAbuf    = (float*)alloc((size_t)B_ * R_ * D_ * 4);
  unsigned short* HP_bf    = (unsigned short*)alloc((size_t)B_ * R_ * C_ * H_ * 2);
  float*          Hf       = (float*)alloc((size_t)B_ * R_ * C_ * D_ * 4);
  unsigned short* H_bf     = (unsigned short*)alloc((size_t)B_ * R_ * C_ * D_ * 2);
  float*          Ef       = (float*)alloc((size_t)B_ * R_ * C_ * D_ * 4);
  float*          groot_f  = (float*)alloc((size_t)B_ * R_ * D_ * 4);
  unsigned short* groot_bf = (unsigned short*)alloc((size_t)B_ * R_ * D_ * 2);
  float*          FAbuf    = (float*)alloc((size_t)B_ * R_ * D_ * 4);

  auto conv = [&](const float* src, unsigned short* dst, int n) {
    k_f2bf<<<(n + 255) / 256, 256, 0, stream>>>(src, dst, n);
  };
  auto gemm = [&](const unsigned short* A, int lda, const unsigned short* Wp, int ldb,
                  const float* bias, const float* resid, float* oF, unsigned short* oH,
                  int M, int N, int K, int act) {
    dim3 g(M / 64, N / 64);
    if (K == 256)
      k_gemm_bf16<256><<<g, 256, 0, stream>>>(A, lda, Wp, ldb, bias, resid, oF, oH, N, act);
    else if (K == 512)
      k_gemm_bf16<512><<<g, 256, 0, stream>>>(A, lda, Wp, ldb, bias, resid, oF, oH, N, act);
    else
      k_gemm_bf16<768><<<g, 256, 0, stream>>>(A, lda, Wp, ldb, bias, resid, oF, oH, N, act);
  };

  // weight conversion
  conv(emb, emb_bf, 50 * D_);
  conv(W1r, W1r_bf, L_ * 3 * D_ * H_);
  conv(W2r, W2r_bf, L_ * H_ * D_);
  conv(Wa,  Wa_bf,  L_ * 2 * D_ * D_);
  conv(W1c, W1c_bf, L_ * 3 * D_ * H_);
  conv(W2c, W2c_bf, L_ * H_ * D_);
  conv(Wf,  Wf_bf,  L_ * D_ * D_);
  conv(Wfa, Wfa_bf, D_ * D_);

  // pooled root/child vectors + edge gather
  k_pooled<<<B_ * R_, 128, 0, stream>>>(att, lstm, rmask, root_f, root_bf, R_);
  k_pooled<<<B_ * C_, 128, 0, stream>>>(att, lstm, cmask, child_f, (unsigned short*)nullptr, C_);
  k_edge_fill<<<B_ * C_, 256, 0, stream>>>(labels, emb_bf, ce_bf);

  for (int i = 0; i < L_; ++i) {
    const unsigned short* W1ri = W1r_bf + (size_t)i * 3 * D_ * H_;
    const unsigned short* W2ri = W2r_bf + (size_t)i * H_ * D_;
    const unsigned short* Wai  = Wa_bf  + (size_t)i * 2 * D_ * D_;
    const unsigned short* W1ci = W1c_bf + (size_t)i * 3 * D_ * H_;
    const unsigned short* W2ci = W2c_bf + (size_t)i * H_ * D_;
    const unsigned short* Wfi  = Wf_bf  + (size_t)i * D_ * D_;

    // child_edge = concat(child_vecs, edge_vecs)
    k_child_refresh<<<B_ * C_, 256, 0, stream>>>(child_f, ce_bf);

    // decomposed first matmul: Rpart = root @ W1r[0:D] + b1r ; CEpart = child_edge @ W1r[D:3D]
    gemm(root_bf, D_, W1ri, H_, b1r + i * H_, nullptr, Rpart, nullptr,
         B_ * R_, H_, D_, 0);
    gemm(ce_bf, 2 * D_, W1ri + (size_t)D_ * H_, H_, nullptr, nullptr, CEpart, nullptr,
         B_ * C_, H_, 2 * D_, 0);
    // RA = root @ Wa[D:2D] + ba  (root_exp half of the e-projection)
    gemm(root_bf, D_, Wai + (size_t)D_ * D_, D_, ba + i * D_, nullptr, RAbuf, nullptr,
         B_ * R_, D_, D_, 0);

    // HP = tanh(Rpart[b,r] + CEpart[b,c])  -> bf16 [B*R*C, H]
    k_hp<<<B_ * R_ * C_, 256, 0, stream>>>(Rpart, CEpart, HP_bf);

    // H = HP @ W2r + b2r ; E = H @ Wa[0:D]
    gemm(HP_bf, H_, W2ri, D_, b2r + i * D_, nullptr, Hf, H_bf,
         B_ * R_ * C_, D_, H_, 0);
    gemm(H_bf, D_, Wai, D_, nullptr, nullptr, Ef, nullptr,
         B_ * R_ * C_, D_, D_, 0);

    // scores / softmax / g_root
    k_gatt<<<B_ * R_, 256, 0, stream>>>(Hf, Ef, RAbuf, gmask, groot_f, groot_bf);

    // child branch: gc = tanh(concat(root[gidx], child_edge) @ W1c + b1c) @ W2c + b2c
    k_gcin<<<B_ * C_, 256, 0, stream>>>(root_bf, ce_bf, gidx, gcin_bf);
    gemm(gcin_bf, 3 * D_, W1ci, H_, b1c + i * H_, nullptr, nullptr, T1_bf,
         B_ * C_, H_, 3 * D_, 1);
    gemm(T1_bf, H_, W2ci, D_, b2c + i * D_, nullptr, nullptr, GC_bf,
         B_ * C_, D_, H_, 0);

    // updates: child += gc @ Wf + bf ; root += g_root @ Wf + bf
    gemm(GC_bf, D_, Wfi, D_, bfv + i * D_, child_f, child_f, nullptr,
         B_ * C_, D_, D_, 0);
    gemm(groot_bf, D_, Wfi, D_, bfv + i * D_, root_f, root_f, nullptr,
         B_ * R_, D_, D_, 0);
    conv(root_f, root_bf, B_ * R_ * D_);
  }

  // fa = tanh(root @ Wfa + bfa); final masked softmax over R
  gemm(root_bf, D_, Wfa_bf, D_, bfa, nullptr, FAbuf, nullptr, B_ * R_, D_, D_, 1);
  k_final<<<B_, 256, 0, stream>>>(FAbuf, root_f, smask, outp);
}
